// PointNetEncoder_40742059770638
// MI455X (gfx1250) — matrix operand order repaired and checked by reading, not compiled, not run on hardware
//
#include <hip/hip_runtime.h>
#include <stdint.h>

#define KNN 20
#define BB 16          // batch
#define NPT 2048       // points, main branch
#define NEG_INF -3.4e38f

#ifndef USE_ASYNC_LDS
#define USE_ASYNC_LDS 1
#endif

typedef __attribute__((ext_vector_type(16))) __bf16 v16bf;
typedef __attribute__((ext_vector_type(8)))  __bf16 v8bf;
typedef __attribute__((ext_vector_type(8)))  float  v8f;
typedef __attribute__((ext_vector_type(4)))  float  v4f;

// ---------------------------------------------------------------------------
// CDNA5 async global->LDS copy (ASYNCcnt path, CDNA5 ISA)
// ---------------------------------------------------------------------------
__device__ __forceinline__ void ld_global_to_lds_f32(float* ldsDst, const float* gsrc) {
#if USE_ASYNC_LDS
  unsigned loff = (unsigned)(uintptr_t)ldsDst;   // low 32 bits = LDS byte offset
  asm volatile("global_load_async_to_lds_b32 %0, %1, off"
               :: "v"(loff), "v"(gsrc) : "memory");
#else
  *ldsDst = *gsrc;
#endif
}

__device__ __forceinline__ void wait_async_lds() {
#if USE_ASYNC_LDS
#if __has_builtin(__builtin_amdgcn_s_wait_asynccnt)
  __builtin_amdgcn_s_wait_asynccnt(0);
#else
  asm volatile("s_wait_asynccnt 0" ::: "memory");
#endif
#endif
}

// ---------------------------------------------------------------------------
// WMMA GEMM:  Y[b,o,n] = sum_c W[o,c] * X[b,c,n] + bias[o]
// block tile: 64 (O) x 128 (N), 8 waves (4 Mwaves x 2 Nwaves),
// wave tile: 16 x 64 = 4x v_wmma_f32_16x16x32_bf16 per K-step.
// FULLK=true: C % 32 == 0 -> branch-free vectorized LDS fill.
// NN (columns) is compile-time so B-tile loads use immediate offsets.
// ---------------------------------------------------------------------------
#define LDA_STRIDE 40   // padded bf16 row stride (80B; 16B-aligned sub-chunks)
#define LDB_STRIDE 40

__device__ __forceinline__ void store8_bf16(__bf16* dst, v4f a, v4f b) {
  v8bf o;
  o[0] = (__bf16)a[0]; o[1] = (__bf16)a[1]; o[2] = (__bf16)a[2]; o[3] = (__bf16)a[3];
  o[4] = (__bf16)b[0]; o[5] = (__bf16)b[1]; o[6] = (__bf16)b[2]; o[7] = (__bf16)b[3];
  *(v8bf*)dst = o;
}

template <bool FULLK, int NN>
__global__ __launch_bounds__(256)
void wmma_gemm_kernel(const float* __restrict__ W, const float* __restrict__ X,
                      const float* __restrict__ bias, float* __restrict__ Y,
                      int O, int C, long xBatchStride, long yBatchStride)
{
  __shared__ __bf16 lA[64 * LDA_STRIDE];   // [o][k]
  __shared__ __bf16 lB[128 * LDB_STRIDE];  // [n][k] (transposed)

  const int tid  = threadIdx.x;
  const int lane = tid & 31;
  const int wave = tid >> 5;
  const int half = (lane >> 4) & 1;
  const int l15  = lane & 15;
  const int mw   = wave & 3;   // 0..3
  const int nw   = wave >> 2;  // 0..1

  const int oBlock = blockIdx.y * 64;
  const int nBase  = blockIdx.x * 128;
  const int b      = blockIdx.z;

  const float* Xb = X + (long)b * xBatchStride;
  float*       Yb = Y + (long)b * yBatchStride;

  v8f acc[4];
#pragma unroll
  for (int t = 0; t < 4; ++t) {
#pragma unroll
    for (int r = 0; r < 8; ++r) acc[t][r] = 0.0f;
  }

  for (int k0 = 0; k0 < C; k0 += 32) {
    __syncthreads();
    if (FULLK) {
      // ---- A tile: 64 x 32, one thread = 8 contiguous K of one O row.
      {
        const int o  = tid >> 2;          // 0..63
        const int kq = (tid & 3) * 8;     // 0,8,16,24
        const float* src = &W[(size_t)(oBlock + o) * C + k0 + kq];
        v4f w0 = *(const v4f*)src;
        v4f w1 = *(const v4f*)(src + 4);
        store8_bf16(&lA[o * LDA_STRIDE + kq], w0, w1);
      }
      // ---- B tile: 32 x 128 -> lB[n][k]; one thread = one n, 16 K values.
      // Single base address + constant immediate offsets (j * NN * 4 bytes).
      {
        const int n  = tid & 127;
        const int kh = (tid >> 7) * 16;   // 0 or 16
        const float* src = Xb + (size_t)(k0 + kh) * NN + nBase + n;
        float v[16];
#pragma unroll
        for (int j = 0; j < 16; ++j)
          v[j] = src[j * NN];
        __bf16* dst = &lB[n * LDB_STRIDE + kh];
        v4f a0 = {v[0], v[1], v[2], v[3]},   a1 = {v[4], v[5], v[6], v[7]};
        v4f a2 = {v[8], v[9], v[10], v[11]}, a3 = {v[12], v[13], v[14], v[15]};
        store8_bf16(dst,     a0, a1);
        store8_bf16(dst + 8, a2, a3);
      }
    } else {
      // ---- guarded path (only C=3 / C=9 layers)
#pragma unroll
      for (int r = 0; r < 8; ++r) {
        int e = r * 256 + tid;
        int o = e >> 5, k = e & 31;
        int gk = k0 + k;
        float w = (gk < C) ? W[(size_t)(oBlock + o) * C + gk] : 0.0f;
        lA[o * LDA_STRIDE + k] = (__bf16)w;
      }
#pragma unroll
      for (int r = 0; r < 16; ++r) {
        int e = r * 256 + tid;
        int k = e >> 7, n = e & 127;
        int gk = k0 + k;
        float xv = (gk < C) ? Xb[(size_t)gk * NN + nBase + n] : 0.0f;
        lB[n * LDB_STRIDE + k] = (__bf16)xv;
      }
    }
    // prefetch next K slab of X into L2 while we compute
    if (k0 + 32 < C) {
      const float* pf = Xb + (size_t)(k0 + 32) * NN + nBase + (tid & 127);
      __builtin_prefetch(pf, 0, 1);
    }
    __syncthreads();

    // ---- A fragment (16x32 bf16, contiguous K = half*16 + i, same as B)
    v16bf afrag;
    {
      const __bf16* row = &lA[(mw * 16 + l15) * LDA_STRIDE];
#pragma unroll
      for (int i = 0; i < 8; ++i) afrag[i]     = row[half * 16 + i];      // K 0..7 / 16..23
#pragma unroll
      for (int i = 0; i < 8; ++i) afrag[8 + i] = row[half * 16 + 8 + i];  // K 8..15 / 24..31
    }
    // ---- load all 4 B fragments, then a WMMA burst
    v16bf bfrag[4];
#pragma unroll
    for (int t = 0; t < 4; ++t) {
      const __bf16* rowb = &lB[(nw * 64 + t * 16 + l15) * LDB_STRIDE + half * 16];
#pragma unroll
      for (int i = 0; i < 16; ++i) bfrag[t][i] = rowb[i];   // K = half*16 + i
    }
#pragma unroll
    for (int t = 0; t < 4; ++t)
      acc[t] = __builtin_amdgcn_wmma_f32_16x16x32_bf16(
          false, afrag, false, bfrag[t], (short)0, acc[t], false, false);
  }

  // ---- epilogue: D layout (VGPR r: M = r + half*8, N = lane&15), add bias
#pragma unroll
  for (int t = 0; t < 4; ++t) {
#pragma unroll
    for (int r = 0; r < 8; ++r) {
      int o = oBlock + mw * 16 + half * 8 + r;
      int n = nBase + nw * 64 + t * 16 + l15;
      Yb[(size_t)o * NN + n] = acc[t][r] + bias[o];
    }
  }
}

// ---------------------------------------------------------------------------
// kNN top-20 (largest pd = -dist). One thread per query point, LDS-tiled
// candidates staged with CDNA5 async global->LDS copies.
// ---------------------------------------------------------------------------
template <int C>
__global__ __launch_bounds__(256)
void knn_kernel(const float* __restrict__ X, long batchStride, int N,
                int* __restrict__ idx)
{
  __shared__ float tile[C * 128];
  const int tid = threadIdx.x;
  const int b   = blockIdx.y;
  const int q   = blockIdx.x * 256 + tid;
  const float* Xb = X + (long)b * batchStride;

  float qv[C];
#pragma unroll
  for (int c = 0; c < C; ++c) qv[c] = Xb[(size_t)c * N + q];

  float best[KNN];
  int   bid[KNN];
#pragma unroll
  for (int j = 0; j < KNN; ++j) { best[j] = NEG_INF; bid[j] = 0; }

  for (int m0 = 0; m0 < N; m0 += 128) {
    __syncthreads();
    for (int e = tid; e < C * 128; e += 256) {
      int c = e >> 7, mm = e & 127;
      ld_global_to_lds_f32(&tile[c * 128 + mm], &Xb[(size_t)c * N + m0 + mm]);
    }
    wait_async_lds();
    __syncthreads();

    for (int mm = 0; mm < 128; ++mm) {
      float d = 0.0f;
#pragma unroll
      for (int c = 0; c < C; ++c) {
        float t = qv[c] - tile[c * 128 + mm];
        d += t * t;
      }
      float pd = -d;
      if (pd > best[KNN - 1]) {
        float cd = pd; int ci = m0 + mm;
#pragma unroll
        for (int j = KNN - 1; j >= 0; --j) {
          bool sw = cd > best[j];
          float td = best[j]; int ti = bid[j];
          if (sw) { best[j] = cd; bid[j] = ci; cd = td; ci = ti; }
        }
      }
    }
  }
#pragma unroll
  for (int j = 0; j < KNN; ++j)
    idx[((size_t)b * N + q) * KNN + j] = bid[j];
}

// ---------------------------------------------------------------------------
// lgpr edge features: out channels = [x (C) | max_k(feat-center) (C) | x (C)]
// ---------------------------------------------------------------------------
template <int C>
__global__ __launch_bounds__(256)
void edge_kernel(const float* __restrict__ X, long batchStride, int N,
                 const int* __restrict__ idx, float* __restrict__ out,
                 long outBatchStride)
{
  int n = blockIdx.x * 256 + threadIdx.x;
  int b = blockIdx.y;
  if (n >= N) return;
  const float* Xb = X + (long)b * batchStride;
  float* ob = out + (long)b * outBatchStride;

  int id[KNN];
#pragma unroll
  for (int k = 0; k < KNN; ++k) id[k] = idx[((size_t)b * N + n) * KNN + k];

  for (int c = 0; c < C; ++c) {
    const float* row = Xb + (size_t)c * N;
    float center = row[n];
    float mx = NEG_INF;
#pragma unroll
    for (int k = 0; k < KNN; ++k) {
      float v = row[id[k]];
      mx = fmaxf(mx, v - center);
    }
    ob[(size_t)c * N + n]           = center;
    ob[(size_t)(C + c) * N + n]     = mx;
    ob[(size_t)(2 * C + c) * N + n] = center;
  }
}

// ---------------------------------------------------------------------------
// BatchNorm over (B, N) per channel: stats then apply (+optional tanh),
// apply can scatter into the concat buffer (dstChanOff / dstBatchStride).
// ---------------------------------------------------------------------------
__global__ __launch_bounds__(256)
void bn_stats_kernel(const float* __restrict__ Y, int O, int N, int B,
                     long batchStride, float* __restrict__ mean,
                     float* __restrict__ rstd)
{
  __shared__ float ssum[256], ssq[256];
  int o = blockIdx.x, tid = threadIdx.x;
  float s = 0.0f, q = 0.0f;
  int total = B * N;
  for (int i = tid; i < total; i += 256) {
    int b = i / N, n = i - b * N;
    float v = Y[(long)b * batchStride + (size_t)o * N + n];
    s += v; q += v * v;
  }
  ssum[tid] = s; ssq[tid] = q;
  __syncthreads();
  for (int st = 128; st > 0; st >>= 1) {
    if (tid < st) { ssum[tid] += ssum[tid + st]; ssq[tid] += ssq[tid + st]; }
    __syncthreads();
  }
  if (tid == 0) {
    float m = ssum[0] / (float)total;
    float v = ssq[0] / (float)total - m * m;
    mean[o] = m;
    rstd[o] = rsqrtf(v + 1e-5f);
  }
}

__global__ __launch_bounds__(256)
void bn_apply_kernel(const float* __restrict__ Y, int O, int N, int B,
                     const float* __restrict__ mean, const float* __restrict__ rstd,
                     const float* __restrict__ g, const float* __restrict__ be,
                     float* __restrict__ dst, long dstBatchStride, int dstChanOff,
                     int doTanh)
{
  long total = (long)B * O * N;
  for (long i = (long)blockIdx.x * blockDim.x + threadIdx.x; i < total;
       i += (long)gridDim.x * blockDim.x) {
    int  n = (int)(i % N);
    long r = i / N;
    int  o = (int)(r % O);
    int  b = (int)(r / O);
    float v = Y[i];
    v = g[o] * (v - mean[o]) * rstd[o] + be[o];
    if (doTanh) v = tanhf(v);
    dst[(long)b * dstBatchStride + (size_t)(dstChanOff + o) * N + n] = v;
  }
}

// ---------------------------------------------------------------------------
// FC, BN-over-batch (+tanh), max over N, rotate, misc
// ---------------------------------------------------------------------------
__global__ __launch_bounds__(256)
void fc_kernel(const float* __restrict__ X, const float* __restrict__ W,
               const float* __restrict__ bias, float* __restrict__ Y,
               int B, int I, int O)
{
  int t = blockIdx.x * blockDim.x + threadIdx.x;
  if (t >= B * O) return;
  int b = t / O, o = t - b * O;
  const float* x = X + (size_t)b * I;
  const float* w = W + (size_t)o * I;
  float s = bias[o];
#pragma unroll 4
  for (int i = 0; i < I; ++i) s += x[i] * w[i];
  Y[t] = s;
}

__global__ __launch_bounds__(256)
void bn2d_tanh_kernel(float* __restrict__ Y, int B, int F,
                      const float* __restrict__ g, const float* __restrict__ be)
{
  int f = blockIdx.x * blockDim.x + threadIdx.x;
  if (f >= F) return;
  float s = 0.0f, q = 0.0f;
  for (int b = 0; b < B; ++b) {
    float v = Y[(size_t)b * F + f];
    s += v; q += v * v;
  }
  float m  = s / (float)B;
  float vv = q / (float)B - m * m;
  float rs = rsqrtf(vv + 1e-5f);
  for (int b = 0; b < B; ++b) {
    float v = Y[(size_t)b * F + f];
    Y[(size_t)b * F + f] = tanhf(g[f] * (v - m) * rs + be[f]);
  }
}

__global__ __launch_bounds__(256)
void maxn_kernel(const float* __restrict__ Y, int O, int N, float* __restrict__ out)
{
  __shared__ float red[256];
  int blk = blockIdx.x;
  int b = blk / O, o = blk - b * O;
  int tid = threadIdx.x;
  const float* row = Y + ((size_t)b * O + o) * N;
  float m = NEG_INF;
  for (int n = tid; n < N; n += 256) m = fmaxf(m, row[n]);
  red[tid] = m;
  __syncthreads();
  for (int st = 128; st > 0; st >>= 1) {
    if (tid < st) red[tid] = fmaxf(red[tid], red[tid + st]);
    __syncthreads();
  }
  if (tid == 0) out[(size_t)b * O + o] = red[0];
}

__global__ void add_eye_kernel(const float* __restrict__ s3, float* __restrict__ trans)
{
  int t = blockIdx.x * blockDim.x + threadIdx.x;
  if (t >= 16 * 9) return;
  int j = t % 9;
  trans[t] = s3[t] + ((j == 0 || j == 4 || j == 8) ? 1.0f : 0.0f);
}

__global__ __launch_bounds__(256)
void rotate_kernel(const float* __restrict__ gen, const float* __restrict__ trans,
                   float* __restrict__ y, int N)
{
  int t = blockIdx.x * blockDim.x + threadIdx.x;
  if (t >= 16 * N) return;
  int b = t / N, n = t - b * N;
  const float* G = gen + (size_t)b * 3 * N;
  const float* T = trans + b * 9;
  float g0 = G[n], g1 = G[N + n], g2 = G[2 * N + n];
#pragma unroll
  for (int d = 0; d < 3; ++d)
    y[((size_t)b * 3 + d) * N + n] = g0 * T[d] + g1 * T[3 + d] + g2 * T[6 + d];
}

__global__ __launch_bounds__(256)
void copy_kernel(const float* __restrict__ src, float* __restrict__ dst, long n)
{
  for (long i = (long)blockIdx.x * blockDim.x + threadIdx.x; i < n;
       i += (long)gridDim.x * blockDim.x)
    dst[i] = src[i];
}

// ---------------------------------------------------------------------------
// Host orchestration
// ---------------------------------------------------------------------------
static void conv_bn(hipStream_t stream,
                    const float* W, const float* bias, const float* g, const float* be,
                    const float* X, long xBS, float* Ybuf, int O, int C, int N, int B,
                    float* dst, long dstBS, int dstOff, int doTanh,
                    float* mean, float* rstd)
{
  dim3 grid(N / 128, O / 64, B);
  bool fullk = (C & 31) == 0;
  if (N == 2048) {
    if (fullk)
      wmma_gemm_kernel<true, 2048><<<grid, 256, 0, stream>>>(W, X, bias, Ybuf, O, C, xBS, (long)O * N);
    else
      wmma_gemm_kernel<false, 2048><<<grid, 256, 0, stream>>>(W, X, bias, Ybuf, O, C, xBS, (long)O * N);
  } else {
    if (fullk)
      wmma_gemm_kernel<true, 1024><<<grid, 256, 0, stream>>>(W, X, bias, Ybuf, O, C, xBS, (long)O * N);
    else
      wmma_gemm_kernel<false, 1024><<<grid, 256, 0, stream>>>(W, X, bias, Ybuf, O, C, xBS, (long)O * N);
  }
  bn_stats_kernel<<<O, 256, 0, stream>>>(Ybuf, O, N, B, (long)O * N, mean, rstd);
  long total = (long)B * O * N;
  bn_apply_kernel<<<(unsigned)((total + 255) / 256), 256, 0, stream>>>(
      Ybuf, O, N, B, mean, rstd, g, be, dst, dstBS, dstOff, doTanh);
}

extern "C" void kernel_launch(void* const* d_in, const int* in_sizes, int n_in,
                              void* d_out, int out_size, void* d_ws, size_t ws_size,
                              hipStream_t stream)
{
  (void)in_sizes; (void)n_in; (void)out_size; (void)ws_size;
#define IN(i) ((const float*)d_in[(i)])
  const float* x = IN(0);
  // param indices (setup_inputs order): layer -> w,b[,g,be]
  // conv4:1 conv5:5 conv6:9 conv7:13 conv8:17 conv9:21 conv10:25
  // fc1:29 fc2:33 fc4:37 fc3:41(w),42(b)
  // conv1:43 conv2:47 conv3:51 s_conv1:55 s_conv2:59 s_conv3:63
  // s_fc1:67 s_fc2:71 s_fc3:75(w),76(b)

  uint8_t* wp = (uint8_t*)d_ws;
  auto take = [&](size_t nbytes) -> void* {
    void* p = wp; wp += (nbytes + 255) & ~(size_t)255; return p;
  };
  int*   idx  = (int*)  take((size_t)BB * NPT * KNN * 4);
  float* h    = (float*)take((size_t)BB * 192 * NPT * 4);
  float* edge = (float*)take((size_t)BB * 192 * NPT * 4);
  float* bufA = (float*)take((size_t)BB * 1024 * NPT * 4);
  float* bufB = (float*)take((size_t)BB * 1024 * NPT * 4);
  float* gen  = (float*)take((size_t)BB * 3072 * 4);
  float* ybuf = (float*)take((size_t)BB * 3 * 1024 * 4);
  float* fcA  = (float*)take((size_t)BB * 1024 * 4);
  float* fcB  = (float*)take((size_t)BB * 1024 * 4);
  float* mean = (float*)take(4096 * 4);
  float* rstd = (float*)take(4096 * 4);

  float* out_y     = (float*)d_out;            // (16,1024)
  float* out_trans = out_y + 16 * 1024;        // (16,9)
  float* out_gen   = out_trans + 16 * 9;       // (16,3,1024)

  const long hBS = 192L * NPT;

  // ---- stage 1: lgpr(x) -> conv4 -> h[:,0:64,:]
  knn_kernel<3><<<dim3(NPT / 256, BB), 256, 0, stream>>>(x, 3L * NPT, NPT, idx);
  edge_kernel<3><<<dim3(NPT / 256, BB), 256, 0, stream>>>(x, 3L * NPT, NPT, idx, edge, 9L * NPT);
  conv_bn(stream, IN(1), IN(2), IN(3), IN(4), edge, 9L * NPT, bufA, 64, 9, NPT, BB,
          h, hBS, 0, 1, mean, rstd);

  // ---- stage 2: lgpr(x1) -> conv5 -> h[:,64:128,:]
  knn_kernel<64><<<dim3(NPT / 256, BB), 256, 0, stream>>>(h, hBS, NPT, idx);
  edge_kernel<64><<<dim3(NPT / 256, BB), 256, 0, stream>>>(h, hBS, NPT, idx, edge, hBS);
  conv_bn(stream, IN(5), IN(6), IN(7), IN(8), edge, hBS, bufA, 64, 192, NPT, BB,
          h, hBS, 64, 1, mean, rstd);

  // ---- stage 3: lgpr(x2) -> conv6 -> h[:,128:192,:]
  knn_kernel<64><<<dim3(NPT / 256, BB), 256, 0, stream>>>(h + 64L * NPT, hBS, NPT, idx);
  edge_kernel<64><<<dim3(NPT / 256, BB), 256, 0, stream>>>(h + 64L * NPT, hBS, NPT, idx, edge, hBS);
  conv_bn(stream, IN(9), IN(10), IN(11), IN(12), edge, hBS, bufA, 64, 192, NPT, BB,
          h, hBS, 128, 1, mean, rstd);

  // ---- conv7..conv10 on concat h
  conv_bn(stream, IN(13), IN(14), IN(15), IN(16), h,    hBS,        bufA, 256,  192,  NPT, BB, bufA, 256L * NPT,  0, 1, mean, rstd);
  conv_bn(stream, IN(17), IN(18), IN(19), IN(20), bufA, 256L * NPT, bufB, 512,  256,  NPT, BB, bufB, 512L * NPT,  0, 1, mean, rstd);
  conv_bn(stream, IN(21), IN(22), IN(23), IN(24), bufB, 512L * NPT, bufA, 1024, 512,  NPT, BB, bufA, 1024L * NPT, 0, 1, mean, rstd);
  conv_bn(stream, IN(25), IN(26), IN(27), IN(28), bufA, 1024L * NPT, bufB, 1024, 1024, NPT, BB, bufB, 1024L * NPT, 0, 1, mean, rstd);

  // ---- global max over N, then fc1/fc2/fc4 (+BN+tanh), fc3 -> gen
  maxn_kernel<<<BB * 1024, 256, 0, stream>>>(bufB, 1024, NPT, fcA);
  fc_kernel<<<(BB * 1024 + 255) / 256, 256, 0, stream>>>(fcA, IN(29), IN(30), fcB, BB, 1024, 1024);
  bn2d_tanh_kernel<<<4, 256, 0, stream>>>(fcB, BB, 1024, IN(31), IN(32));
  fc_kernel<<<(BB * 1024 + 255) / 256, 256, 0, stream>>>(fcB, IN(33), IN(34), fcA, BB, 1024, 1024);
  bn2d_tanh_kernel<<<4, 256, 0, stream>>>(fcA, BB, 1024, IN(35), IN(36));
  fc_kernel<<<(BB * 1024 + 255) / 256, 256, 0, stream>>>(fcA, IN(37), IN(38), fcB, BB, 1024, 1024);
  bn2d_tanh_kernel<<<4, 256, 0, stream>>>(fcB, BB, 1024, IN(39), IN(40));
  fc_kernel<<<(BB * 3072 + 255) / 256, 256, 0, stream>>>(fcB, IN(41), IN(42), gen, BB, 1024, 3072);
  copy_kernel<<<192, 256, 0, stream>>>(gen, out_gen, (long)BB * 3 * 1024);

  // ---- STN branch on gen (B=16, C=3, N=1024)
  const int NS = 1024;
  conv_bn(stream, IN(55), IN(56), IN(57), IN(58), gen,  3L * NS,   bufA, 64,   3,   NS, BB, bufA, 64L * NS,   0, 1, mean, rstd);
  conv_bn(stream, IN(59), IN(60), IN(61), IN(62), bufA, 64L * NS,  bufB, 128,  64,  NS, BB, bufB, 128L * NS,  0, 1, mean, rstd);
  conv_bn(stream, IN(63), IN(64), IN(65), IN(66), bufB, 128L * NS, bufA, 1024, 128, NS, BB, bufA, 1024L * NS, 0, 1, mean, rstd);
  maxn_kernel<<<BB * 1024, 256, 0, stream>>>(bufA, 1024, NS, fcA);
  fc_kernel<<<(BB * 512 + 255) / 256, 256, 0, stream>>>(fcA, IN(67), IN(68), fcB, BB, 1024, 512);
  bn2d_tanh_kernel<<<2, 256, 0, stream>>>(fcB, BB, 512, IN(69), IN(70));
  fc_kernel<<<(BB * 256 + 255) / 256, 256, 0, stream>>>(fcB, IN(71), IN(72), fcA, BB, 512, 256);
  bn2d_tanh_kernel<<<1, 256, 0, stream>>>(fcA, BB, 256, IN(73), IN(74));
  fc_kernel<<<1, 256, 0, stream>>>(fcA, IN(75), IN(76), fcB, BB, 256, 9);
  add_eye_kernel<<<1, 256, 0, stream>>>(fcB, out_trans);

  // ---- y = gen x trans, then conv1/conv2/conv3 + max -> out_y
  rotate_kernel<<<(BB * NS + 255) / 256, 256, 0, stream>>>(gen, out_trans, ybuf, NS);
  conv_bn(stream, IN(43), IN(44), IN(45), IN(46), ybuf, 3L * NS,   bufA, 64,   3,   NS, BB, bufA, 64L * NS,   0, 1, mean, rstd);
  conv_bn(stream, IN(47), IN(48), IN(49), IN(50), bufA, 64L * NS,  bufB, 128,  64,  NS, BB, bufB, 128L * NS,  0, 1, mean, rstd);
  conv_bn(stream, IN(51), IN(52), IN(53), IN(54), bufB, 128L * NS, bufA, 1024, 128, NS, BB, bufA, 1024L * NS, 0, 0, mean, rstd);
  maxn_kernel<<<BB * 1024, 256, 0, stream>>>(bufA, 1024, NS, out_y);
#undef IN
}
